// GCNTopK_41068477284546
// MI455X (gfx1250) — matrix operand order repaired and checked by reading, not compiled, not run on hardware
//
#include <hip/hip_runtime.h>
#include <math.h>

#define BGR   32
#define N0V   1024
#define DEG   8
#define EDGES (BGR * N0V * DEG)   // 262144
#define FIN   300
#define NHID  256
#define NCLS  10

// ---------------------------------------------------------------------------
// WMMA types (gfx1250, wave32)
// ---------------------------------------------------------------------------
typedef __attribute__((ext_vector_type(16))) __bf16          v16bf;
typedef __attribute__((ext_vector_type(8)))  float           v8f;
typedef __attribute__((ext_vector_type(8)))  unsigned short  su8;

union Frag16x32 { su8 u[2]; v16bf v; };

__device__ __forceinline__ unsigned short f2bf_rne(float x) {
  unsigned int u = __builtin_bit_cast(unsigned int, x);
  u += 0x7FFFu + ((u >> 16) & 1u);
  return (unsigned short)(u >> 16);
}

// padded LDS row stride (ushorts): 80 bytes -> 16B aligned, bank-conflict free
#define LDT 40

// assemble the 16x32 A fragment for this wave's row strip
__device__ __forceinline__ v16bf load_a_frag(const unsigned short* __restrict__ lA,
                                             int wave, int l15, int hi) {
  // lanes 0-15 : halves 0-7 -> K 0-7,  halves 8-15 -> K 16-23
  // lanes 16-31: halves 0-7 -> K 8-15, halves 8-15 -> K 24-31
  Frag16x32 af;
  const unsigned short* abase = &lA[((wave << 4) | l15) * LDT];
  af.u[0] = *(const su8*)(abase + (hi << 3));
  af.u[1] = *(const su8*)(abase + (hi << 3) + 16);
  return af.v;
}

// 4 WMMAs across the 64-wide N strip with a shared A fragment
__device__ __forceinline__ void wmma_4(v16bf a, const unsigned short* __restrict__ lB,
                                       int l15, int hi, v8f acc[4]) {
#pragma unroll
  for (int s = 0; s < 4; ++s) {
    // B: lanes 0-15 -> K 0-7,16-23 ; lanes 16-31 -> K 8-15,24-31 ; N = s*16 + l15
    Frag16x32 bfr;
    const unsigned short* bbase = &lB[((s << 4) | l15) * LDT];
    bfr.u[0] = *(const su8*)(bbase + (hi << 3));
    bfr.u[1] = *(const su8*)(bbase + (hi << 3) + 16);
    acc[s] = __builtin_amdgcn_wmma_f32_16x16x32_bf16(
        false, a, false, bfr.v, (short)0, acc[s], false, false);
  }
}

// stage one 32x64 weight tile transposed into LDS (coalesced float4 along N)
__device__ __forceinline__ void stage_w(const float* __restrict__ W, int k0, int col0,
                                        int tid, unsigned short* __restrict__ lB) {
#pragma unroll
  for (int it = 0; it < 4; ++it) {
    int i  = tid + it * 128;           // 0..511 float4 units (32 k x 16 n4)
    int kk = i >> 4;
    int n4 = (i & 15) << 2;
    float4 v = *(const float4*)&W[(size_t)(k0 + kk) * NHID + col0 + n4];
    lB[(n4 + 0) * LDT + kk] = f2bf_rne(v.x);
    lB[(n4 + 1) * LDT + kk] = f2bf_rne(v.y);
    lB[(n4 + 2) * LDT + kk] = f2bf_rne(v.z);
    lB[(n4 + 3) * LDT + kk] = f2bf_rne(v.w);
  }
}

__device__ __forceinline__ void stage_w_tail(const float* __restrict__ W, int kFull,
                                             int rem, int col0, int tid,
                                             unsigned short* __restrict__ lB) {
  for (int i = tid; i < 64 * 32; i += 128) {
    int n = i >> 5, kk = i & 31;
    lB[n * LDT + kk] =
        f2bf_rne(kk < rem ? W[(size_t)(kFull + kk) * NHID + col0 + n] : 0.0f);
  }
}

// ---------------------------------------------------------------------------
// Dual GEMM sharing the A operand:
//   Cr[M,256] = A[M,K] * Wr[K,256]
//   Co[M,256] = A[M,K] * Wo[K,256]
// block = 128 threads (4 waves); block tile 64x64; wave tile 16x64 (x2 outputs).
// grid = (M/64, 256/64).  M % 64 == 0; K = 300 or 256.
// ---------------------------------------------------------------------------
__global__ __launch_bounds__(128)
void gemm_dual_bf16_wmma(const float* __restrict__ A,
                         const float* __restrict__ Wr,
                         const float* __restrict__ Wo,
                         float* __restrict__ Cr, float* __restrict__ Co,
                         int M, int K) {
  __shared__ __align__(16) unsigned short lA[64 * LDT];   // [m][k]
  __shared__ __align__(16) unsigned short lBr[64 * LDT];  // [n][k] transposed
  __shared__ __align__(16) unsigned short lBo[64 * LDT];  // [n][k] transposed
  const int tid  = threadIdx.x;
  const int wave = tid >> 5;
  const int lane = tid & 31;
  const int l15  = lane & 15;
  const int hi   = lane >> 4;          // 0 or 1
  const int row0 = blockIdx.x * 64;
  const int col0 = blockIdx.y * 64;

  v8f accR[4] = {};
  v8f accO[4] = {};

  const int kFull = K & ~31;           // full 32-wide k tiles

  // -------- full tiles: predicate-free vectorized staging --------
  const int arow = tid >> 1;            // 0..63
  const int aks  = (tid & 1) << 4;      // 0 or 16
  for (int k0 = 0; k0 < kFull; k0 += 32) {
    // A tile: 16 contiguous floats per thread -> 4x float4 loads (staged once,
    // consumed by both weight streams)
    {
      const float4* src = (const float4*)&A[(size_t)(row0 + arow) * K + k0 + aks];
      unsigned short tmp[16];
#pragma unroll
      for (int q = 0; q < 4; ++q) {
        float4 v = src[q];
        tmp[4 * q + 0] = f2bf_rne(v.x);
        tmp[4 * q + 1] = f2bf_rne(v.y);
        tmp[4 * q + 2] = f2bf_rne(v.z);
        tmp[4 * q + 3] = f2bf_rne(v.w);
      }
      *(su8*)&lA[arow * LDT + aks]     = *(su8*)&tmp[0];
      *(su8*)&lA[arow * LDT + aks + 8] = *(su8*)&tmp[8];
    }
    stage_w(Wr, k0, col0, tid, lBr);
    stage_w(Wo, k0, col0, tid, lBo);
    __syncthreads();

    // prefetch next A k-tile while the WMMAs run (global_prefetch_b8)
    if (k0 + 32 < K) {
      __builtin_prefetch(&A[(size_t)(row0 + arow) * K + k0 + 32 + aks], 0, 1);
    }

    v16bf a = load_a_frag(lA, wave, l15, hi);
    wmma_4(a, lBr, l15, hi, accR);
    wmma_4(a, lBo, l15, hi, accO);
    __syncthreads();
  }

  // -------- tail tile (runs at most once; only K=300 layer) --------
  if (kFull < K) {
    const int rem = K - kFull;
    for (int i = tid; i < 64 * 32; i += 128) {
      int m = i >> 5, kk = i & 31;
      lA[m * LDT + kk] =
          f2bf_rne(kk < rem ? A[(size_t)(row0 + m) * K + kFull + kk] : 0.0f);
    }
    stage_w_tail(Wr, kFull, rem, col0, tid, lBr);
    stage_w_tail(Wo, kFull, rem, col0, tid, lBo);
    __syncthreads();
    v16bf a = load_a_frag(lA, wave, l15, hi);
    wmma_4(a, lBr, l15, hi, accR);
    wmma_4(a, lBo, l15, hi, accO);
    __syncthreads();
  }

  // C/D layout: VGPR j -> M = j + 8*hi ; N = l15
  const int mbase = row0 + (wave << 4) + (hi << 3);
#pragma unroll
  for (int s = 0; s < 4; ++s) {
    const int c = col0 + (s << 4) + l15;
#pragma unroll
    for (int j = 0; j < 8; ++j) {
      Cr[(size_t)(mbase + j) * NHID + c] = accR[s][j];
      Co[(size_t)(mbase + j) * NHID + c] = accO[s][j];
    }
  }
}

// ---------------------------------------------------------------------------
// helpers
// ---------------------------------------------------------------------------
__global__ void fill_f32(float* p, float v, int n) {
  int i = blockIdx.x * blockDim.x + threadIdx.x;
  if (i < n) p[i] = v;
}

__global__ void edge_init(const int* __restrict__ ei, int* src, int* dst, int* em) {
  int i = blockIdx.x * blockDim.x + threadIdx.x;
  if (i < EDGES) { src[i] = ei[i]; dst[i] = ei[EDGES + i]; em[i] = 1; }
}

// agg[dst] += h_rel[src] over active edges; 4 feats / thread
__global__ void scatter_add(const float* __restrict__ hrel,
                            const int* __restrict__ src,
                            const int* __restrict__ dst,
                            const int* __restrict__ em, float* agg) {
  int i = blockIdx.x * blockDim.x + threadIdx.x;
  if (i >= EDGES * 64) return;
  int e = i >> 6;
  if (!em[e]) return;
  int f = (i & 63) << 2;
  const float4 v = *(const float4*)&hrel[(size_t)src[e] * NHID + f];
  float* a = &agg[(size_t)dst[e] * NHID + f];
  atomicAdd(a + 0, v.x); atomicAdd(a + 1, v.y);
  atomicAdd(a + 2, v.z); atomicAdd(a + 3, v.w);
}

__global__ void combine_relu(const float* __restrict__ agg,
                             const float* __restrict__ hroot,
                             const float* __restrict__ bias,
                             float* out, int rows) {
  int i = blockIdx.x * blockDim.x + threadIdx.x;
  if (i < rows * NHID) {
    float v = agg[i] + hroot[i] + bias[i & (NHID - 1)];
    out[i] = v > 0.0f ? v : 0.0f;
  }
}

__global__ void pnorm_kernel(const float* __restrict__ p, float* out) {
  __shared__ float s[NHID];
  int t = threadIdx.x;
  float v = p[t];
  s[t] = v * v;
  __syncthreads();
  for (int o = NHID / 2; o > 0; o >>= 1) {
    if (t < o) s[t] += s[t + o];
    __syncthreads();
  }
  if (t == 0) out[0] = rsqrtf(s[0]);
}

// one wave per node: score = tanh(dot(x,p) * invnorm)
__global__ void score_kernel(const float* __restrict__ x,
                             const float* __restrict__ p,
                             const float* __restrict__ invn,
                             float* sc, int rows) {
  int gid  = blockIdx.x * blockDim.x + threadIdx.x;
  int wid  = gid >> 5;
  int lane = threadIdx.x & 31;
  if (wid >= rows) return;
  float s = 0.0f;
  for (int f = lane; f < NHID; f += 32) s += x[(size_t)wid * NHID + f] * p[f];
  for (int o = 16; o > 0; o >>= 1) s += __shfl_down(s, o, 32);
  if (lane == 0) sc[wid] = tanhf(s * invn[0]);
}

// per-graph bitonic sort (descending, index tie-break) + top-k selection
__global__ __launch_bounds__(512)
void topk_sort(const float* __restrict__ sc, float* vals, int* gidx, int* m,
               int n, int kkeep) {
  __shared__ float sv[1024];
  __shared__ int   si[1024];
  const int b = blockIdx.x, t = threadIdx.x;
  for (int i = t; i < 1024; i += 512) {
    if (i < n) { sv[i] = sc[b * n + i]; si[i] = i; }
    else       { sv[i] = -INFINITY;     si[i] = 0x7FFFFFFF; }
  }
  for (int i = t; i < n; i += 512) m[b * n + i] = -1;
  __syncthreads();

  for (int ksz = 2; ksz <= 1024; ksz <<= 1) {
    for (int j = ksz >> 1; j > 0; j >>= 1) {
      for (int i = t; i < 1024; i += 512) {
        int ixj = i ^ j;
        if (ixj > i) {
          bool descSeg = ((i & ksz) == 0);
          float va = sv[i], vb = sv[ixj];
          int   ia = si[i], ib = si[ixj];
          bool aFirst = (va > vb) || (va == vb && ia < ib);
          if (descSeg != aFirst) {
            sv[i] = vb; sv[ixj] = va; si[i] = ib; si[ixj] = ia;
          }
        }
      }
      __syncthreads();
    }
  }

  for (int i = t; i < kkeep; i += 512) {
    int loc = si[i];
    vals[b * kkeep + i] = sv[i];
    gidx[b * kkeep + i] = b * n + loc;
    m[b * n + loc]      = b * kkeep + i;
  }
}

__global__ void gather_gate(const float* __restrict__ xin,
                            const int* __restrict__ gidx,
                            const float* __restrict__ vals,
                            float* xout, int rowsOut) {
  int i = blockIdx.x * blockDim.x + threadIdx.x;
  if (i >= rowsOut * 64) return;
  int r = i >> 6, f = (i & 63) << 2;
  float g = vals[r];
  const float4 v = *(const float4*)&xin[(size_t)gidx[r] * NHID + f];
  float4 o; o.x = v.x * g; o.y = v.y * g; o.z = v.z * g; o.w = v.w * g;
  *(float4*)&xout[(size_t)r * NHID + f] = o;
}

__global__ void edge_remap(int* src, int* dst, int* em, const int* __restrict__ m) {
  int i = blockIdx.x * blockDim.x + threadIdx.x;
  if (i >= EDGES) return;
  if (em[i]) {
    int ns = m[src[i]], nd = m[dst[i]];
    int v = (ns >= 0 && nd >= 0) ? 1 : 0;
    src[i] = v ? ns : 0; dst[i] = v ? nd : 0; em[i] = v;
  } else { src[i] = 0; dst[i] = 0; }
}

// per-graph max & mean readout accumulated into total[B, 512]
__global__ void readout_acc(const float* __restrict__ x, float* total, int kkeep) {
  int b = blockIdx.x, f = threadIdx.x;  // 256 threads
  float mx = -INFINITY, sm = 0.0f;
  for (int r = 0; r < kkeep; ++r) {
    float v = x[(size_t)(b * kkeep + r) * NHID + f];
    mx = fmaxf(mx, v); sm += v;
  }
  total[b * (2 * NHID) + f]        += mx;
  total[b * (2 * NHID) + NHID + f] += sm / (float)kkeep;
}

__global__ void mlp_relu(const float* __restrict__ in, const float* __restrict__ W,
                         const float* __restrict__ bias, float* out, int K, int N) {
  int b = blockIdx.x, c = threadIdx.x;
  if (c >= N) return;
  float s = bias[c];
  for (int k = 0; k < K; ++k) s += in[b * K + k] * W[k * N + c];
  out[b * N + c] = fmaxf(s, 0.0f);
}

__global__ void mlp_head(const float* __restrict__ in, const float* __restrict__ W,
                         const float* __restrict__ bias, float* out) {
  int b = blockIdx.x * blockDim.x + threadIdx.x;
  if (b >= BGR) return;
  float lg[NCLS];
  float mx = -INFINITY;
  for (int c = 0; c < NCLS; ++c) {
    float s = bias[c];
    for (int k = 0; k < NHID / 2; ++k) s += in[b * (NHID / 2) + k] * W[k * NCLS + c];
    lg[c] = s; mx = fmaxf(mx, s);
  }
  float se = 0.0f;
  for (int c = 0; c < NCLS; ++c) se += expf(lg[c] - mx);
  float lse = mx + logf(se);
  for (int c = 0; c < NCLS; ++c) {
    out[b * NCLS + c] = lg[c] - lse;                               // log_softmax
    out[BGR * NCLS + b * NCLS + c] = 1.0f / (1.0f + expf(-lg[c])); // sigmoid
  }
}

// ---------------------------------------------------------------------------
// driver
// ---------------------------------------------------------------------------
extern "C" void kernel_launch(void* const* d_in, const int* in_sizes, int n_in,
                              void* d_out, int out_size, void* d_ws, size_t ws_size,
                              hipStream_t stream) {
  (void)in_sizes; (void)n_in; (void)out_size; (void)ws_size;

  const float* x0 = (const float*)d_in[0];
  const int*   ei = (const int*)d_in[1];
  const float* Wrel[4]  = {(const float*)d_in[2],  (const float*)d_in[6],
                           (const float*)d_in[10], (const float*)d_in[14]};
  const float* brel[4]  = {(const float*)d_in[3],  (const float*)d_in[7],
                           (const float*)d_in[11], (const float*)d_in[15]};
  const float* Wroot[4] = {(const float*)d_in[4],  (const float*)d_in[8],
                           (const float*)d_in[12], (const float*)d_in[16]};
  const float* pv[4]    = {(const float*)d_in[5],  (const float*)d_in[9],
                           (const float*)d_in[13], (const float*)d_in[17]};
  const float* lin1W = (const float*)d_in[18]; const float* lin1b = (const float*)d_in[19];
  const float* lin2W = (const float*)d_in[20]; const float* lin2b = (const float*)d_in[21];
  const float* lin3W = (const float*)d_in[22]; const float* lin3b = (const float*)d_in[23];

  char* ws = (char*)d_ws;
  size_t off = 0;
  auto alloc = [&](size_t bytes) -> void* {
    void* p = ws + off;
    off = (off + bytes + 255) & ~(size_t)255;
    return p;
  };

  const size_t BIG = (size_t)BGR * N0V * NHID * sizeof(float);  // 33.5 MB
  float* bufX    = (float*)alloc(BIG);
  float* bufRel  = (float*)alloc(BIG);
  float* bufRoot = (float*)alloc(BIG);
  float* bufAgg  = (float*)alloc(BIG);
  int*   srcE    = (int*)alloc(EDGES * sizeof(int));
  int*   dstE    = (int*)alloc(EDGES * sizeof(int));
  int*   emE     = (int*)alloc(EDGES * sizeof(int));
  float* scores  = (float*)alloc((size_t)BGR * N0V * sizeof(float));
  float* vals    = (float*)alloc((size_t)BGR * 768 * sizeof(float));
  int*   gidx    = (int*)alloc((size_t)BGR * 768 * sizeof(int));
  int*   mmap    = (int*)alloc((size_t)BGR * N0V * sizeof(int));
  float* pnorm   = (float*)alloc(256);
  float* total   = (float*)alloc((size_t)BGR * 2 * NHID * sizeof(float));
  float* h1      = (float*)alloc((size_t)BGR * NHID * sizeof(float));
  float* h2      = (float*)alloc((size_t)BGR * (NHID / 2) * sizeof(float));

  const int nL[4] = {1024, 768, 576, 432};
  const int kL[4] = {768, 576, 432, 324};

  fill_f32<<<(BGR * 2 * NHID + 255) / 256, 256, 0, stream>>>(total, 0.0f, BGR * 2 * NHID);
  edge_init<<<EDGES / 256, 256, 0, stream>>>(ei, srcE, dstE, emE);

  const float* xin = x0;
  int din = FIN;
  for (int l = 0; l < 4; ++l) {
    const int rowsIn  = BGR * nL[l];
    const int rowsOut = BGR * kL[l];
    dim3 gg(rowsIn / 64, NHID / 64);

    // fused dual GEMM: stages A once for both W_rel and W_root streams
    gemm_dual_bf16_wmma<<<gg, 128, 0, stream>>>(xin, Wrel[l], Wroot[l],
                                                bufRel, bufRoot, rowsIn, din);

    fill_f32<<<((size_t)rowsIn * NHID + 255) / 256, 256, 0, stream>>>(
        bufAgg, 0.0f, rowsIn * NHID);
    scatter_add<<<(EDGES * 64) / 256, 256, 0, stream>>>(bufRel, srcE, dstE, emE, bufAgg);
    combine_relu<<<((size_t)rowsIn * NHID + 255) / 256, 256, 0, stream>>>(
        bufAgg, bufRoot, brel[l], bufRoot, rowsIn);

    pnorm_kernel<<<1, NHID, 0, stream>>>(pv[l], pnorm);
    score_kernel<<<(rowsIn * 32 + 255) / 256, 256, 0, stream>>>(
        bufRoot, pv[l], pnorm, scores, rowsIn);
    topk_sort<<<BGR, 512, 0, stream>>>(scores, vals, gidx, mmap, nL[l], kL[l]);

    gather_gate<<<((size_t)rowsOut * 64 + 255) / 256, 256, 0, stream>>>(
        bufRoot, gidx, vals, bufX, rowsOut);
    readout_acc<<<BGR, NHID, 0, stream>>>(bufX, total, kL[l]);
    edge_remap<<<EDGES / 256, 256, 0, stream>>>(srcE, dstE, emE, mmap);

    xin = bufX;
    din = NHID;
  }

  mlp_relu<<<BGR, NHID, 0, stream>>>(total, lin1W, lin1b, h1, 2 * NHID, NHID);
  mlp_relu<<<BGR, NHID / 2, 0, stream>>>(h1, lin2W, lin2b, h2, NHID, NHID / 2);
  mlp_head<<<1, BGR, 0, stream>>>(h2, lin3W, lin3b, (float*)d_out);
}
